// RSSMCore_67138928771117
// MI455X (gfx1250) — compile-verified, matrix operand order checked by reading, not run on hardware
//
#include <hip/hip_runtime.h>
#include <hip/hip_bf16.h>

// ---------------------------------------------------------------------------
// RSSM core for MI455X (gfx1250): bf16 WMMA GEMMs, L2-resident weights.
// Weights pre-transposed to [N][K] bf16 so both A and B WMMA fragments load
// from LDS as two ds_load_b128 each. Global->LDS staging uses the CDNA5
// async-to-LDS path (ASYNCcnt) when the toolchain exposes it.
// ---------------------------------------------------------------------------

typedef __attribute__((ext_vector_type(16))) __bf16 v16bf;
typedef __attribute__((ext_vector_type(8)))  float  v8f;

#define GRU_H 2048
#define NCAT  32
#define NCLS  32
#define ZF    1024
#define MLP_P 1024   // 1000 padded
#define ADIM  6
#define FDIM  1024
#define T_LEN 64
#define B_SZ  256
#define XIN_K 1088   // 1024 + 6 padded to 64-multiple (BK | K for all GEMMs)
#define G3H   6144
#define HF_K  3072

#if defined(__HIP_DEVICE_COMPILE__)
#if __has_builtin(__builtin_amdgcn_global_load_async_to_lds_b128) && \
    __has_builtin(__builtin_amdgcn_s_wait_asynccnt)
#define USE_ASYNC_LDS 1
#endif
#endif

#ifdef USE_ASYNC_LDS
#define AS1 __attribute__((address_space(1)))
#define AS3 __attribute__((address_space(3)))
typedef int v4i_ __attribute__((vector_size(16)));   // matches builtin prototype
__device__ __forceinline__ void async_cp_b128(const unsigned short* g,
                                              unsigned short* l) {
    // GLOBAL_LOAD_ASYNC_TO_LDS_B128: per-lane global -> LDS, ASYNCcnt tracked
    // prototype: (int4 as1* gsrc, int4 as3* ldst, imm offset, imm cpol)
    __builtin_amdgcn_global_load_async_to_lds_b128((AS1 v4i_*)g, (AS3 v4i_*)l, 0, 0);
}
#endif

__device__ __forceinline__ unsigned short f32_to_bf16(float f) {
    union { float f; unsigned u; } x; x.f = f;
    unsigned r = x.u + 0x7FFFu + ((x.u >> 16) & 1u);   // round-to-nearest-even
    if ((x.u & 0x7F800000u) == 0x7F800000u) r = x.u;   // inf/nan passthrough
    return (unsigned short)(r >> 16);
}

// ---------------------------------------------------------------------------
// Tiled bf16 WMMA GEMM:  Out[M,N] = act( A[M,K](bf16) @ WT[N,K](bf16)^T + bias )
// Block tile 32(M) x 128(N), K-chunk 64, double-buffered LDS.
// 256 threads = 8 wave32s arranged 2(M) x 4(N); each wave owns a 16x32 strip:
// per K-chunk it issues 4 x v_wmma_f32_16x16x32_bf16 fed by 12 ds_load_b128.
// All dims padded to tile multiples -> no bounds checks anywhere.
// ---------------------------------------------------------------------------
#define BM 32
#define BN 128
#define BK 64
#define LDK 72   // LDS row stride (ushorts): multiple of 8 -> 16B-aligned rows

__global__ __launch_bounds__(256)
void gemm_bf16_wmma(const unsigned short* __restrict__ A, int lda,
                    const unsigned short* __restrict__ WT, int ldk,
                    const float* __restrict__ bias,
                    void* __restrict__ Out, int ldo,
                    int K, int actElu, int outBf16)
{
    __shared__ __align__(16) unsigned short As[2][BM][LDK];
    __shared__ __align__(16) unsigned short Bs[2][BN][LDK];

    const int tid  = threadIdx.x;
    const int lane = tid & 31;
    const int wave = tid >> 5;
    const int wm   = wave >> 2;     // 0..1  (M sub-tile)
    const int wn   = wave & 3;      // 0..3  (N sub-tile)
    const int m0   = blockIdx.y * BM;
    const int n0   = blockIdx.x * BN;

    // cooperative staging indices (everything 16B aligned)
    const int am  = tid >> 3;           // A row 0..31
    const int ak  = (tid & 7) * 8;      // A col 0..56 (one b128 per thread)
    const int bn  = tid >> 1;           // WT row 0..127
    const int bkq = (tid & 1) * 32;     // WT col 0/32 (four b128 per thread)

    const unsigned short* aSrc = A  + (size_t)(m0 + am) * lda + ak;
    const unsigned short* bSrc = WT + (size_t)(n0 + bn) * ldk + bkq;

    // fragment addressing (per documented 16-bit WMMA layouts):
    // element e of the v16bf frag maps to K = kbase + e + (e & 8), with
    // kbase = sub*32 + 8*(lane>=16)  -> two contiguous 16B LDS reads per frag.
    const int mrow  = wm * 16 + (lane & 15);
    const int nrow0 = wn * 32 + (lane & 15);
    const int khalf = (lane >> 4) * 8;

    v8f c0 = {};
    v8f c1 = {};

    auto compute = [&](int cb) {
        #pragma unroll
        for (int s = 0; s < 2; ++s) {
            const int kb = s * 32 + khalf;
            union { v16bf v; uint4 q[2]; } af, b0, b1;
            af.q[0] = *(const uint4*)&As[cb][mrow][kb];
            af.q[1] = *(const uint4*)&As[cb][mrow][kb + 16];
            b0.q[0] = *(const uint4*)&Bs[cb][nrow0][kb];
            b0.q[1] = *(const uint4*)&Bs[cb][nrow0][kb + 16];
            b1.q[0] = *(const uint4*)&Bs[cb][nrow0 + 16][kb];
            b1.q[1] = *(const uint4*)&Bs[cb][nrow0 + 16][kb + 16];
            c0 = __builtin_amdgcn_wmma_f32_16x16x32_bf16(false, af.v, false, b0.v,
                                                         (short)0, c0, false, false);
            c1 = __builtin_amdgcn_wmma_f32_16x16x32_bf16(false, af.v, false, b1.v,
                                                         (short)0, c1, false, false);
        }
    };

#ifdef USE_ASYNC_LDS
    // --- CDNA5 async global->LDS staging (no VGPR round-trip) ---
    auto stage = [&](int buf, int koff) {
        async_cp_b128(aSrc + koff,      &As[buf][am][ak]);
        async_cp_b128(bSrc + koff,      &Bs[buf][bn][bkq]);
        async_cp_b128(bSrc + koff + 8,  &Bs[buf][bn][bkq + 8]);
        async_cp_b128(bSrc + koff + 16, &Bs[buf][bn][bkq + 16]);
        async_cp_b128(bSrc + koff + 24, &Bs[buf][bn][bkq + 24]);
    };
    stage(0, 0);
    __builtin_amdgcn_s_wait_asynccnt(0);
    __syncthreads();
    int cur = 0;
    for (int k0 = 0; k0 < K; k0 += BK) {
        const bool more = (k0 + BK) < K;
        if (more) stage(cur ^ 1, k0 + BK);   // DMA next chunk behind the WMMAs
        compute(cur);
        if (more) {
            __builtin_amdgcn_s_wait_asynccnt(0);
            __syncthreads();
            cur ^= 1;
        }
    }
#else
    // --- fallback: register-staged double buffering ---
    uint4 aR  = *(const uint4*)(aSrc);
    uint4 bR0 = *(const uint4*)(bSrc);
    uint4 bR1 = *(const uint4*)(bSrc + 8);
    uint4 bR2 = *(const uint4*)(bSrc + 16);
    uint4 bR3 = *(const uint4*)(bSrc + 24);
    int cur = 0;
    *(uint4*)&As[0][am][ak]       = aR;
    *(uint4*)&Bs[0][bn][bkq]      = bR0;
    *(uint4*)&Bs[0][bn][bkq + 8]  = bR1;
    *(uint4*)&Bs[0][bn][bkq + 16] = bR2;
    *(uint4*)&Bs[0][bn][bkq + 24] = bR3;
    __syncthreads();

    for (int k0 = 0; k0 < K; k0 += BK) {
        const bool more = (k0 + BK) < K;
        if (more) {
            const unsigned short* a2 = aSrc + k0 + BK;
            const unsigned short* b2 = bSrc + k0 + BK;
            aR  = *(const uint4*)(a2);
            bR0 = *(const uint4*)(b2);
            bR1 = *(const uint4*)(b2 + 8);
            bR2 = *(const uint4*)(b2 + 16);
            bR3 = *(const uint4*)(b2 + 24);
        }
        compute(cur);
        if (more) {
            const int nxt = cur ^ 1;
            *(uint4*)&As[nxt][am][ak]       = aR;
            *(uint4*)&Bs[nxt][bn][bkq]      = bR0;
            *(uint4*)&Bs[nxt][bn][bkq + 8]  = bR1;
            *(uint4*)&Bs[nxt][bn][bkq + 16] = bR2;
            *(uint4*)&Bs[nxt][bn][bkq + 24] = bR3;
            cur = nxt;
            __syncthreads();
        }
    }
#endif

    // Epilogue: C/D layout — VGPR r holds M = r + 8*(lane>=16), N = lane&15.
    const int nc    = n0 + wn * 32 + (lane & 15);
    const int mbase = m0 + wm * 16 + 8 * (lane >> 4);
    #pragma unroll
    for (int r = 0; r < 8; ++r) {
        const int m = mbase + r;
        float v0 = c0[r] + bias[nc];
        float v1 = c1[r] + bias[nc + 16];
        if (actElu) {
            v0 = v0 > 0.f ? v0 : (__expf(v0) - 1.f);
            v1 = v1 > 0.f ? v1 : (__expf(v1) - 1.f);
        }
        if (outBf16) {
            unsigned short* o = (unsigned short*)Out;
            o[(size_t)m * ldo + nc]      = f32_to_bf16(v0);
            o[(size_t)m * ldo + nc + 16] = f32_to_bf16(v1);
        } else {
            float* o = (float*)Out;
            o[(size_t)m * ldo + nc]      = v0;
            o[(size_t)m * ldo + nc + 16] = v1;
        }
    }
}

// ---------------------------------------------------------------------------
// Weight repack: fp32 [Ks,Ns] -> transposed bf16 [Nd][Kd], zero-padded.
// ---------------------------------------------------------------------------
__global__ void pack_w(const float* __restrict__ src, int Ks, int Ns,
                       unsigned short* __restrict__ dst, int Kd, int Nd)
{
    size_t i = (size_t)blockIdx.x * 256 + threadIdx.x;
    size_t tot = (size_t)Kd * Nd;
    if (i >= tot) return;
    int k = (int)(i % Kd);
    int n = (int)(i / Kd);
    float v = (k < Ks && n < Ns) ? src[(size_t)k * Ns + n] : 0.f;
    dst[i] = f32_to_bf16(v);
}

__global__ void pack_bias(const float* __restrict__ src, int Ns,
                          float* __restrict__ dst, int Nd)
{
    int i = blockIdx.x * 256 + threadIdx.x;
    if (i < Nd) dst[i] = (i < Ns) ? src[i] : 0.f;
}

// xin = [z(1024) | a_onehot(6) | pad(58)] per row; init: zeros, action=onehot(0)
__global__ void init_xin(unsigned short* __restrict__ xin)
{
    int i = blockIdx.x * 256 + threadIdx.x;   // 256*1088 total
    xin[i] = ((i % XIN_K) == ZF) ? (unsigned short)0x3F80 : (unsigned short)0;
}

__global__ void init_state(float* __restrict__ h, unsigned short* __restrict__ h_b)
{
    int i = blockIdx.x * 256 + threadIdx.x;   // 256*2048 total
    h[i] = 0.f;
    h_b[i] = 0;
}

// ---------------------------------------------------------------------------
// GRU gate fusion (torch GRUCell order r,z,n) + bf16 mirror of h into h_b and
// the first 2048 cols of the [h|feat] posterior input.
// ---------------------------------------------------------------------------
__global__ __launch_bounds__(256)
void gru_update(const float* __restrict__ gi, const float* __restrict__ gh,
                float* __restrict__ h,
                unsigned short* __restrict__ h_b,
                unsigned short* __restrict__ hf_b)
{
    const int b = blockIdx.x;
    const float* gib = gi + (size_t)b * G3H;
    const float* ghb = gh + (size_t)b * G3H;
    for (int i = threadIdx.x; i < GRU_H; i += 256) {
        float ir = gib[i], iz = gib[GRU_H + i], in = gib[2 * GRU_H + i];
        float hr = ghb[i], hz = ghb[GRU_H + i], hn = ghb[2 * GRU_H + i];
        float r  = 1.f / (1.f + __expf(-(ir + hr)));
        float zg = 1.f / (1.f + __expf(-(iz + hz)));
        float n  = tanhf(in + r * hn);
        float hv = h[(size_t)b * GRU_H + i];
        float hnew = (1.f - zg) * n + zg * hv;
        h[(size_t)b * GRU_H + i] = hnew;
        unsigned short hb = f32_to_bf16(hnew);
        h_b[(size_t)b * GRU_H + i] = hb;
        hf_b[(size_t)b * HF_K + i] = hb;
    }
}

__global__ void feat_to_hf(const float* __restrict__ feats,
                           unsigned short* __restrict__ hf_b, int t)
{
    const int b = blockIdx.x;
    const float* f = feats + ((size_t)t * B_SZ + b) * FDIM;
    for (int j = threadIdx.x; j < FDIM; j += 256)
        hf_b[(size_t)b * HF_K + GRU_H + j] = f32_to_bf16(f[j]);
}

// ---------------------------------------------------------------------------
// Posterior softmax + categorical sample; straight-through z == one-hot.
// One thread per (batch, category) distribution of 32 classes.
// ---------------------------------------------------------------------------
__global__ void sample_z(const float* __restrict__ post,
                         unsigned short* __restrict__ xin, int t)
{
    int g = blockIdx.x * 256 + threadIdx.x;      // 0..8191
    int b = g >> 5, cat = g & 31;
    const float* lg = post + (size_t)b * ZF + cat * NCLS;
    float mx = lg[0];
    #pragma unroll
    for (int i = 1; i < NCLS; ++i) mx = fmaxf(mx, lg[i]);
    float p[NCLS], s = 0.f;
    #pragma unroll
    for (int i = 0; i < NCLS; ++i) { p[i] = __expf(lg[i] - mx); s += p[i]; }
    unsigned x = (unsigned)(t * 9781 + g * 6271) + 0x9E3779B9u;
    x ^= x >> 16; x *= 0x7FEB352Du; x ^= x >> 15; x *= 0x846CA68Bu; x ^= x >> 16;
    float u = (x >> 8) * (1.f / 16777216.f) * s;
    int idx = 0; float acc = 0.f;
    #pragma unroll
    for (int i = 0; i < NCLS; ++i) { acc += p[i]; if (u >= acc) idx = i + 1; }
    if (idx > NCLS - 1) idx = NCLS - 1;
    unsigned short* zrow = xin + (size_t)b * XIN_K + cat * NCLS;
    #pragma unroll
    for (int i = 0; i < NCLS; ++i)
        zrow[i] = (i == idx) ? (unsigned short)0x3F80 : (unsigned short)0;
}

__global__ void set_action(const int* __restrict__ actions,
                           unsigned short* __restrict__ xin, int t)
{
    int b = threadIdx.x;  // 256 threads
    int a = actions[(size_t)t * B_SZ + b];
    unsigned short* arow = xin + (size_t)b * XIN_K + ZF;
    #pragma unroll
    for (int i = 0; i < ADIM; ++i)
        arow[i] = (i == a) ? (unsigned short)0x3F80 : (unsigned short)0;
}

// ---------------------------------------------------------------------------
extern "C" void kernel_launch(void* const* d_in, const int* in_sizes, int n_in,
                              void* d_out, int out_size, void* d_ws, size_t ws_size,
                              hipStream_t stream)
{
    (void)in_sizes; (void)n_in; (void)out_size; (void)ws_size;

    const float* features = (const float*)d_in[0];
    const int*   actions  = (const int*)d_in[1];
    const float* W_in = (const float*)d_in[2];
    const float* b_in = (const float*)d_in[3];
    const float* W_ih = (const float*)d_in[4];
    const float* W_hh = (const float*)d_in[5];
    const float* b_ih = (const float*)d_in[6];
    const float* b_hh = (const float*)d_in[7];
    const float* Wp1  = (const float*)d_in[8];
    const float* bp1  = (const float*)d_in[9];
    const float* Wp2  = (const float*)d_in[10];
    const float* bp2  = (const float*)d_in[11];
    const float* Wq1  = (const float*)d_in[12];
    const float* bq1  = (const float*)d_in[13];
    const float* Wq2  = (const float*)d_in[14];
    const float* bq2  = (const float*)d_in[15];

    // bump allocator over d_ws (256B aligned slices)
    char* ws = (char*)d_ws;
    size_t off = 0;
    auto alloc = [&](size_t bytes) -> char* {
        char* p = ws + off;
        off = (off + bytes + 255) & ~(size_t)255;
        return p;
    };

    // bf16 transposed weights [N][K], padded; L2-resident across all 64 steps
    unsigned short* wb_in = (unsigned short*)alloc((size_t)MLP_P * XIN_K * 2);
    unsigned short* wb_ih = (unsigned short*)alloc((size_t)G3H * MLP_P * 2);
    unsigned short* wb_hh = (unsigned short*)alloc((size_t)G3H * GRU_H * 2);
    unsigned short* wb_p1 = (unsigned short*)alloc((size_t)MLP_P * GRU_H * 2);
    unsigned short* wb_p2 = (unsigned short*)alloc((size_t)ZF * MLP_P * 2);
    unsigned short* wb_q1 = (unsigned short*)alloc((size_t)MLP_P * HF_K * 2);
    unsigned short* wb_q2 = (unsigned short*)alloc((size_t)ZF * MLP_P * 2);
    float* bb_in = (float*)alloc(MLP_P * 4);
    float* bb_ih = (float*)alloc(G3H * 4);
    float* bb_hh = (float*)alloc(G3H * 4);
    float* bb_p1 = (float*)alloc(MLP_P * 4);
    float* bb_p2 = (float*)alloc(ZF * 4);
    float* bb_q1 = (float*)alloc(MLP_P * 4);
    float* bb_q2 = (float*)alloc(ZF * 4);
    // activations
    unsigned short* xin  = (unsigned short*)alloc((size_t)B_SZ * XIN_K * 2);
    unsigned short* x_b  = (unsigned short*)alloc((size_t)B_SZ * MLP_P * 2);
    float*          h_f  = (float*)alloc((size_t)B_SZ * GRU_H * 4);
    unsigned short* h_b  = (unsigned short*)alloc((size_t)B_SZ * GRU_H * 2);
    unsigned short* hf_b = (unsigned short*)alloc((size_t)B_SZ * HF_K * 2);
    float*          gi_f = (float*)alloc((size_t)B_SZ * G3H * 4);
    float*          gh_f = (float*)alloc((size_t)B_SZ * G3H * 4);
    unsigned short* p1_b = (unsigned short*)alloc((size_t)B_SZ * MLP_P * 2);
    unsigned short* q1_b = (unsigned short*)alloc((size_t)B_SZ * MLP_P * 2);

    float* outPrior = (float*)d_out;                          // [64,256,1024]
    float* outPost  = outPrior + (size_t)T_LEN * B_SZ * ZF;   // [64,256,1024]

    auto gridFor = [](size_t tot) { return dim3((unsigned)((tot + 255) / 256)); };

    // ---- one-time per launch: repack weights/biases (deterministic) ----
    pack_w<<<gridFor((size_t)MLP_P * XIN_K), 256, 0, stream>>>(W_in, ZF + ADIM, 1000, wb_in, XIN_K, MLP_P);
    pack_w<<<gridFor((size_t)G3H * MLP_P), 256, 0, stream>>>(W_ih, 1000, G3H, wb_ih, MLP_P, G3H);
    pack_w<<<gridFor((size_t)G3H * GRU_H), 256, 0, stream>>>(W_hh, GRU_H, G3H, wb_hh, GRU_H, G3H);
    pack_w<<<gridFor((size_t)MLP_P * GRU_H), 256, 0, stream>>>(Wp1, GRU_H, 1000, wb_p1, GRU_H, MLP_P);
    pack_w<<<gridFor((size_t)ZF * MLP_P), 256, 0, stream>>>(Wp2, 1000, ZF, wb_p2, MLP_P, ZF);
    pack_w<<<gridFor((size_t)MLP_P * HF_K), 256, 0, stream>>>(Wq1, HF_K, 1000, wb_q1, HF_K, MLP_P);
    pack_w<<<gridFor((size_t)ZF * MLP_P), 256, 0, stream>>>(Wq2, 1000, ZF, wb_q2, MLP_P, ZF);
    pack_bias<<<gridFor(MLP_P), 256, 0, stream>>>(b_in, 1000, bb_in, MLP_P);
    pack_bias<<<gridFor(G3H), 256, 0, stream>>>(b_ih, G3H, bb_ih, G3H);
    pack_bias<<<gridFor(G3H), 256, 0, stream>>>(b_hh, G3H, bb_hh, G3H);
    pack_bias<<<gridFor(MLP_P), 256, 0, stream>>>(bp1, 1000, bb_p1, MLP_P);
    pack_bias<<<gridFor(ZF), 256, 0, stream>>>(bp2, ZF, bb_p2, ZF);
    pack_bias<<<gridFor(MLP_P), 256, 0, stream>>>(bq1, 1000, bb_q1, MLP_P);
    pack_bias<<<gridFor(ZF), 256, 0, stream>>>(bq2, ZF, bb_q2, ZF);
    init_xin<<<gridFor((size_t)B_SZ * XIN_K), 256, 0, stream>>>(xin);
    init_state<<<gridFor((size_t)B_SZ * GRU_H), 256, 0, stream>>>(h_f, h_b);

    const dim3 blk(256);
    const dim3 gSmall(MLP_P / BN, B_SZ / BM);   // N=1024 GEMMs: 8 x 8
    const dim3 gGate(G3H / BN, B_SZ / BM);      // N=6144 GEMMs: 48 x 8

    // ---- 64-step sequential scan ----
    for (int t = 0; t < T_LEN; ++t) {
        // x = elu([z|a] @ W_in + b_in)            -> bf16
        gemm_bf16_wmma<<<gSmall, blk, 0, stream>>>(xin, XIN_K, wb_in, XIN_K, bb_in,
                                                   x_b, MLP_P, XIN_K, 1, 1);
        // gi = x @ W_ih + b_ih                    -> f32
        gemm_bf16_wmma<<<gGate, blk, 0, stream>>>(x_b, MLP_P, wb_ih, MLP_P, bb_ih,
                                                  gi_f, G3H, MLP_P, 0, 0);
        // gh = h @ W_hh + b_hh                    -> f32
        gemm_bf16_wmma<<<gGate, blk, 0, stream>>>(h_b, GRU_H, wb_hh, GRU_H, bb_hh,
                                                  gh_f, G3H, GRU_H, 0, 0);
        // h = GRU(gi, gh, h); mirrors into h_b and hf_b[:, :2048]
        gru_update<<<B_SZ, blk, 0, stream>>>(gi_f, gh_f, h_f, h_b, hf_b);
        // hf_b[:, 2048:] = bf16(features[t])
        feat_to_hf<<<B_SZ, blk, 0, stream>>>(features, hf_b, t);
        // prior = elu(h @ Wp1 + bp1) @ Wp2 + bp2  -> d_out
        gemm_bf16_wmma<<<gSmall, blk, 0, stream>>>(h_b, GRU_H, wb_p1, GRU_H, bb_p1,
                                                   p1_b, MLP_P, GRU_H, 1, 1);
        gemm_bf16_wmma<<<gSmall, blk, 0, stream>>>(p1_b, MLP_P, wb_p2, MLP_P, bb_p2,
                                                   outPrior + (size_t)t * B_SZ * ZF, ZF,
                                                   MLP_P, 0, 0);
        // post = elu([h|feat] @ Wq1 + bq1) @ Wq2 + bq2 -> d_out
        gemm_bf16_wmma<<<gSmall, blk, 0, stream>>>(hf_b, HF_K, wb_q1, HF_K, bb_q1,
                                                   q1_b, MLP_P, HF_K, 1, 1);
        gemm_bf16_wmma<<<gSmall, blk, 0, stream>>>(q1_b, MLP_P, wb_q2, MLP_P, bb_q2,
                                                   outPost + (size_t)t * B_SZ * ZF, ZF,
                                                   MLP_P, 0, 0);
        // sample z_{t+1} (straight-through == one-hot) and a_onehot(actions[t])
        if (t < T_LEN - 1) {
            sample_z<<<dim3(32), blk, 0, stream>>>(outPost + (size_t)t * B_SZ * ZF, xin, t);
            set_action<<<dim3(1), blk, 0, stream>>>(actions, xin, t);
        }
    }
}
